// CharBiLSTMCRF_1202590843433
// MI455X (gfx1250) — matrix operand order repaired and checked by reading, not compiled, not run on hardware
//
#include <hip/hip_runtime.h>
#include <hip/hip_bf16.h>

// ============================ problem constants ============================
#define Bsz   64
#define Ssz   128
#define Wsz   16
#define EMBD  300
#define HID   256
#define CEMBD 50
#define CHID  100
#define NTAGS 50
#define NSEQ  (Bsz*Ssz)          // 8192 word positions
#define NCHAR (NSEQ*Wsz)         // 131072 char positions
#define CKP   64                 // char-LSTM input K padded (50 -> 64)
#define CHKP  128                // char-LSTM hidden K padded (100 -> 128)
#define XK    416                // word-LSTM input K padded (400 -> 416)
#define WG4   (4*HID)            // 1024 word gate columns
#define CG4   (4*CHID)           // 400 char gate columns

typedef __attribute__((ext_vector_type(16))) __bf16 v16bf;
typedef __attribute__((ext_vector_type(8)))  float  v8f;

#define DEV __device__ __forceinline__

// ============================ scalar helpers ============================
DEV unsigned short f32_to_bf16_bits(float f) {
  unsigned int u = __float_as_uint(f);
  unsigned int r = u + 0x7FFFu + ((u >> 16) & 1u);   // round-to-nearest-even
  return (unsigned short)(r >> 16);
}
DEV __bf16 f32_to_bf16(float f) {
  union { unsigned short u; __bf16 h; } cv;
  cv.u = f32_to_bf16_bits(f);
  return cv.h;
}
DEV float sigmoidf_(float x) { return 1.0f / (1.0f + __expf(-x)); }

// ============================ WMMA fragment helpers ============================
// A fragment: 16(M) x 32(K) bf16 tile from row-major memory [.. , ldk].
// Per ISA 7.12.2: lane L -> M = L%16, half = L/16; elements 0..7 at K = half*8+e,
// elements 8..15 at K = 16+half*8+e  -> two contiguous 16B chunks -> b128 loads.
DEV v16bf load_a_frag_g(const __bf16* base, int ldk, int k0, int lane) {
  int m = lane & 15, half = lane >> 4;
  const __bf16* p = base + (size_t)m * ldk + k0 + half * 8;
  union { uint4 q; __bf16 e[8]; } lo, hi;
  lo.q = *(const uint4*)p;
  hi.q = *(const uint4*)(p + 16);
  v16bf a;
#pragma unroll
  for (int e = 0; e < 8; ++e) { a[e] = lo.e[e]; a[8 + e] = hi.e[e]; }
  return a;
}

// Same A-fragment pattern but from bf16 h-state in LDS -> two ds_load_b128, no cvt.
DEV v16bf load_a_frag_lds(const __bf16* h, int ldk, int k0, int lane) {
  int m = lane & 15, half = lane >> 4;
  const __bf16* p = h + m * ldk + k0 + half * 8;
  union { uint4 q; __bf16 e[8]; } lo, hi;
  lo.q = *(const uint4*)p;
  hi.q = *(const uint4*)(p + 16);
  v16bf a;
#pragma unroll
  for (int e = 0; e < 8; ++e) { a[e] = lo.e[e]; a[8 + e] = hi.e[e]; }
  return a;
}

// B fragment: 32(K) x 16(N) bf16. Weights stored "K-contiguous": [Ncols][ldk].
// lane L -> N = L%16, elements 0..7: K = (L/16)*8 + e, 8..15: K = 16 + (L/16)*8 + e.
DEV v16bf load_b_frag_g(const __bf16* baseN0, int ldk, int k0, int lane) {
  int n = lane & 15, hi2 = lane >> 4;
  const __bf16* p = baseN0 + (size_t)n * ldk + k0 + hi2 * 8;
  union { uint4 q; __bf16 e[8]; } lo, hi;
  lo.q = *(const uint4*)p;
  hi.q = *(const uint4*)(p + 16);
  v16bf b;
#pragma unroll
  for (int e = 0; e < 8; ++e) { b[e] = lo.e[e]; b[8 + e] = hi.e[e]; }
  return b;
}

DEV v8f wmma_bf16(v16bf a, v16bf b, v8f c) {
  return __builtin_amdgcn_wmma_f32_16x16x32_bf16(false, a, false, b, (short)0, c, false, false);
}

// ============================ weight prep kernels ============================
// Gate-interleave + bf16 + K-pad: dst[4u+g][k] = src[g*H+u][k] (k<D else 0).
__global__ void conv_gate_w_kernel(const float* __restrict__ src, __bf16* __restrict__ dst,
                                   int H, int D, int Dpad) {
  long total = (long)4 * H * Dpad;
  long idx = (long)blockIdx.x * blockDim.x + threadIdx.x;
  if (idx >= total) return;
  int k = (int)(idx % Dpad);
  int rp = (int)(idx / Dpad);
  int u = rp >> 2, g = rp & 3;
  float v = (k < D) ? src[(size_t)(g * H + u) * D + k] : 0.0f;
  dst[idx] = f32_to_bf16(v);
}

// Plain bf16 convert with zero row padding: dst [Npad][D].
__global__ void conv_plain_w_kernel(const float* __restrict__ src, __bf16* __restrict__ dst,
                                    int Nsrc, int Npad, int D) {
  long total = (long)Npad * D;
  long idx = (long)blockIdx.x * blockDim.x + threadIdx.x;
  if (idx >= total) return;
  int k = (int)(idx % D);
  int n = (int)(idx / D);
  float v = (n < Nsrc) ? src[(size_t)n * D + k] : 0.0f;
  dst[idx] = f32_to_bf16(v);
}

__global__ void perm_bias_kernel(const float* __restrict__ src, float* __restrict__ dst, int H) {
  int idx = blockIdx.x * blockDim.x + threadIdx.x;
  if (idx >= 4 * H) return;
  int u = idx >> 2, g = idx & 3;
  dst[idx] = src[g * H + u];
}

// ============================ char embedding gather ============================
// ce bf16 [NCHAR][CKP], zero padded K.
__global__ void ce_gather_kernel(const int* __restrict__ words, const float* __restrict__ char_emb,
                                 __bf16* __restrict__ ce) {
  long idx = (long)blockIdx.x * blockDim.x + threadIdx.x;
  if (idx >= (long)NCHAR * CKP) return;
  int k = (int)(idx & (CKP - 1));
  long i = idx >> 6;
  float v = 0.0f;
  if (k < CEMBD) {
    int id = words[i];
    v = char_emb[(size_t)id * CEMBD + k];
  }
  ce[idx] = f32_to_bf16(v);
}

// ============================ generic WMMA GEMM ============================
// C[M][ldc] (f32) = A(bf16 [M][ldk]) x B(bf16 [N][ldk], K-contiguous) (+ bias[n<biasN])
__global__ __launch_bounds__(256)
void gemm_bf16_kernel(const __bf16* __restrict__ A, const __bf16* __restrict__ B,
                      float* __restrict__ C, const float* __restrict__ bias, int biasN,
                      int ldk, int ldc, int nKtiles, int nNtiles) {
  int lane = threadIdx.x & 31;
  int wave = threadIdx.x >> 5;
  int wavesPerBlock = blockDim.x >> 5;
  int mt = blockIdx.x;
  int jt = blockIdx.y * wavesPerBlock + wave;
  if (jt >= nNtiles) return;                     // wave-uniform, EXEC stays full
  const __bf16* Abase = A + (size_t)mt * 16 * ldk;
  const __bf16* Bbase = B + (size_t)jt * 16 * ldk;
  v8f acc = {};
  for (int kt = 0; kt < nKtiles; ++kt) {
    __builtin_prefetch(Bbase + (size_t)(kt + 1) * 32, 0, 1);   // global_prefetch_b8
    v16bf a = load_a_frag_g(Abase, ldk, kt * 32, lane);
    v16bf b = load_b_frag_g(Bbase, ldk, kt * 32, lane);
    acc = wmma_bf16(a, b, acc);
  }
  int n = jt * 16 + (lane & 15);
  float bn = (bias && n < biasN) ? bias[n] : 0.0f;
  int mrow = mt * 16 + (lane >> 4) * 8;
#pragma unroll
  for (int v = 0; v < 8; ++v)
    C[(size_t)(mrow + v) * ldc + n] = acc[v] + bn;
}

// ============================ char LSTM recurrence ============================
// Persistent: block owns 16 of the 8192 char sequences, runs all 16 steps.
// Gate layout is unit-interleaved (col' = 4u+g) -> cell update is wave-local.
// h-state lives in LDS as bf16 (WMMA-operand precision), ping-ponged per step.
#define CH_NT 25        // 400 gate cols / 16
#define CH_WAVES 4
__global__ __launch_bounds__(128)
void char_lstm_kernel(const __bf16* __restrict__ ce,     // [NCHAR][CKP]
                      const __bf16* __restrict__ Wi,     // [CG4][CKP]  (permuted)
                      const __bf16* __restrict__ Wh,     // [CG4][CHKP] (permuted)
                      const float*  __restrict__ bias,   // [CG4] (permuted)
                      float* __restrict__ hout,          // [NSEQ][CHID] final h
                      int reverse) {
  __shared__ __align__(16) __bf16 hlds[2][16][CHKP];   // ping-pong h (bf16)
  __shared__ float scr[CH_WAVES][16][16];              // per-wave gate tile bounce
  int lane = threadIdx.x & 31;
  int wave = threadIdx.x >> 5;
  int r0 = blockIdx.x * 16;
  for (int i = threadIdx.x; i < 2 * 16 * CHKP / 2; i += blockDim.x)
    ((unsigned int*)&hlds[0][0][0])[i] = 0u;           // bf16 zero bits
  float creg[7][2];
#pragma unroll
  for (int a2 = 0; a2 < 7; ++a2) { creg[a2][0] = 0.0f; creg[a2][1] = 0.0f; }
  __syncthreads();
  int cur = 0;
#pragma unroll 1
  for (int step = 0; step < Wsz; ++step) {
    int t = reverse ? (Wsz - 1 - step) : step;
    const __bf16* ceb = ce + ((size_t)r0 * Wsz + t) * CKP;   // row stride Wsz*CKP
    // Hoist A-fragments (invariant over the N-tile loop): 2 input + 4 hidden.
    v16bf ax[CKP / 32], ah[CHKP / 32];
#pragma unroll
    for (int kt = 0; kt < CKP / 32; ++kt)
      ax[kt] = load_a_frag_g(ceb, Wsz * CKP, kt * 32, lane);
#pragma unroll
    for (int kt = 0; kt < CHKP / 32; ++kt)
      ah[kt] = load_a_frag_lds(&hlds[cur][0][0], CHKP, kt * 32, lane);
#pragma unroll
    for (int jj = 0; jj < 7; ++jj) {
      int j = wave + jj * CH_WAVES;          // N-tile (wave-uniform)
      if (j < CH_NT) {
        v8f acc = {};
        const __bf16* Bi = Wi + (size_t)j * 16 * CKP;
#pragma unroll
        for (int kt = 0; kt < CKP / 32; ++kt)
          acc = wmma_bf16(ax[kt], load_b_frag_g(Bi, CKP, kt * 32, lane), acc);
        const __bf16* Bh = Wh + (size_t)j * 16 * CHKP;
#pragma unroll
        for (int kt = 0; kt < CHKP / 32; ++kt)
          acc = wmma_bf16(ah[kt], load_b_frag_g(Bh, CHKP, kt * 32, lane), acc);
        float bn = bias[j * 16 + (lane & 15)];
#pragma unroll
        for (int v = 0; v < 8; ++v)
          scr[wave][(lane >> 4) * 8 + v][lane & 15] = acc[v] + bn;
        // wave-local LSTM cell: tile cols = 4 complete units (i,f,g,o adjacent)
#pragma unroll
        for (int p = 0; p < 2; ++p) {
          int o = p * 32 + lane;
          int m = o >> 2, lu = o & 3;
          float ig = scr[wave][m][4 * lu + 0];
          float fg = scr[wave][m][4 * lu + 1];
          float gg = scr[wave][m][4 * lu + 2];
          float og = scr[wave][m][4 * lu + 3];
          float cc = sigmoidf_(fg) * creg[jj][p] + sigmoidf_(ig) * tanhf(gg);
          creg[jj][p] = cc;
          float hh = sigmoidf_(og) * tanhf(cc);
          hlds[cur ^ 1][m][j * 4 + lu] = f32_to_bf16(hh);
        }
      }
    }
    __syncthreads();
    cur ^= 1;
  }
  for (int i = threadIdx.x; i < 16 * CHID; i += blockDim.x) {
    int m = i / CHID, u = i % CHID;
    hout[(size_t)(r0 + m) * CHID + u] = (float)hlds[cur][m][u];
  }
}

// ============================ word x / xr assembly ============================
DEV float word_x_val(int b, int t, int k, const int* sentences, const float* word_emb,
                     const float* hf, const float* hb) {
  int r = b * Ssz + t;
  if (k < EMBD)            return word_emb[(size_t)sentences[r] * EMBD + k];
  else if (k < EMBD + CHID) return hf[(size_t)r * CHID + (k - EMBD)] + hb[(size_t)r * CHID + (k - EMBD)];
  return 0.0f;
}
__global__ void x_assembly_kernel(const int* __restrict__ sentences, const int* __restrict__ lengths,
                                  const float* __restrict__ word_emb,
                                  const float* __restrict__ hf, const float* __restrict__ hb,
                                  __bf16* __restrict__ x, __bf16* __restrict__ xr) {
  long idx = (long)blockIdx.x * blockDim.x + threadIdx.x;
  if (idx >= (long)NSEQ * XK) return;
  int k = (int)(idx % XK);
  int r = (int)(idx / XK);
  int b = r >> 7, s = r & (Ssz - 1);
  x[idx] = f32_to_bf16(word_x_val(b, s, k, sentences, word_emb, hf, hb));
  int len = lengths[b];
  float vr = 0.0f;
  if (s < len) vr = word_x_val(b, len - 1 - s, k, sentences, word_emb, hf, hb);
  xr[idx] = f32_to_bf16(vr);
}

// ============================ word LSTM recurrence ============================
// 4 blocks x 16 batch rows, 8 waves each own 8 of the 64 gate N-tiles. 128 steps.
#define WD_WAVES 8
__global__ __launch_bounds__(256)
void word_lstm_kernel(const float* __restrict__ xWi,   // [NSEQ][WG4] (permuted cols)
                      const __bf16* __restrict__ Wh,   // [WG4][HID]  (permuted)
                      const float*  __restrict__ bias, // [WG4] (permuted)
                      float* __restrict__ hall) {      // [NSEQ][HID] all h states
  __shared__ __align__(16) __bf16 hlds[2][16][HID];
  __shared__ float scr[WD_WAVES][16][16];
  int lane = threadIdx.x & 31;
  int wave = threadIdx.x >> 5;
  int b0 = blockIdx.x * 16;
  for (int i = threadIdx.x; i < 2 * 16 * HID / 2; i += blockDim.x)
    ((unsigned int*)&hlds[0][0][0])[i] = 0u;
  float creg[8][2];
#pragma unroll
  for (int a2 = 0; a2 < 8; ++a2) { creg[a2][0] = 0.0f; creg[a2][1] = 0.0f; }
  __syncthreads();
  int cur = 0;
#pragma unroll 1
  for (int t = 0; t < Ssz; ++t) {
    // Hoist the 8 h A-fragments (invariant over the 8 N-tiles this wave owns).
    v16bf ah[HID / 32];
#pragma unroll
    for (int kt = 0; kt < HID / 32; ++kt)
      ah[kt] = load_a_frag_lds(&hlds[cur][0][0], HID, kt * 32, lane);
#pragma unroll
    for (int jj = 0; jj < 8; ++jj) {
      int j = wave * 8 + jj;                  // N-tile 0..63
      v8f acc = {};
      const __bf16* Bh = Wh + (size_t)j * 16 * HID;
#pragma unroll
      for (int kt = 0; kt < HID / 32; ++kt)
        acc = wmma_bf16(ah[kt], load_b_frag_g(Bh, HID, kt * 32, lane), acc);
      int n = j * 16 + (lane & 15);
      float bn = bias[n];
#pragma unroll
      for (int v = 0; v < 8; ++v) {
        int m = (lane >> 4) * 8 + v;
        float base = xWi[((size_t)(b0 + m) * Ssz + t) * WG4 + n];
        scr[wave][m][lane & 15] = acc[v] + base + bn;
      }
#pragma unroll
      for (int p = 0; p < 2; ++p) {
        int o = p * 32 + lane;
        int m = o >> 2, lu = o & 3;
        float ig = scr[wave][m][4 * lu + 0];
        float fg = scr[wave][m][4 * lu + 1];
        float gg = scr[wave][m][4 * lu + 2];
        float og = scr[wave][m][4 * lu + 3];
        float cc = sigmoidf_(fg) * creg[jj][p] + sigmoidf_(ig) * tanhf(gg);
        creg[jj][p] = cc;
        float hh = sigmoidf_(og) * tanhf(cc);
        int u = j * 4 + lu;
        hlds[cur ^ 1][m][u] = f32_to_bf16(hh);
        hall[((size_t)(b0 + m) * Ssz + t) * HID + u] = hh;
      }
    }
    __syncthreads();
    cur ^= 1;
  }
}

// ============================ out assembly ============================
// out[b,s] = valid ? concat(h_fwd[b,s], hr[b, len-1-s]) : 0   (bf16 [NSEQ][2H])
__global__ void out_assembly_kernel(const int* __restrict__ lengths,
                                    const float* __restrict__ hf_all, const float* __restrict__ hr_all,
                                    __bf16* __restrict__ out) {
  long idx = (long)blockIdx.x * blockDim.x + threadIdx.x;
  if (idx >= (long)NSEQ * 2 * HID) return;
  int k = (int)(idx & (2 * HID - 1));
  int r = (int)(idx >> 9);
  int b = r >> 7, s = r & (Ssz - 1);
  int len = lengths[b];
  float v = 0.0f;
  if (s < len) {
    if (k < HID) v = hf_all[(size_t)r * HID + k];
    else         v = hr_all[((size_t)b * Ssz + (len - 1 - s)) * HID + (k - HID)];
  }
  out[idx] = f32_to_bf16(v);
}

// ============================ CRF ============================
#define ELD 64   // emit leading dim (NTAGS padded)
__global__ __launch_bounds__(64)
void crf_kernel(const float* __restrict__ emit,     // [NSEQ][ELD]
                const int* __restrict__ lengths, const int* __restrict__ tags,
                const float* __restrict__ trans, float* __restrict__ outv) {
  __shared__ float tr[NTAGS * NTAGS];
  __shared__ float alpha[NTAGS];
  __shared__ float nalpha[NTAGS];
  int b = blockIdx.x;
  int j = threadIdx.x;
  for (int i = j; i < NTAGS * NTAGS; i += 64) tr[i] = trans[i];
  int len = lengths[b];
  __syncthreads();
  if (j < NTAGS) alpha[j] = emit[((size_t)b * Ssz + 0) * ELD + j];
  __syncthreads();
  for (int t = 1; t < Ssz; ++t) {
    if (t < len && j < NTAGS) {
      float mx = -1e30f;
      for (int i = 0; i < NTAGS; ++i) mx = fmaxf(mx, alpha[i] + tr[i * NTAGS + j]);
      float s = 0.0f;
      for (int i = 0; i < NTAGS; ++i) s += __expf(alpha[i] + tr[i * NTAGS + j] - mx);
      nalpha[j] = mx + __logf(s) + emit[((size_t)b * Ssz + t) * ELD + j];
    }
    __syncthreads();
    if (t < len && j < NTAGS) alpha[j] = nalpha[j];
    __syncthreads();
  }
  if (j == 0) {
    float mx = -1e30f;
    for (int i = 0; i < NTAGS; ++i) mx = fmaxf(mx, alpha[i]);
    float s = 0.0f;
    for (int i = 0; i < NTAGS; ++i) s += __expf(alpha[i] - mx);
    float pred = mx + __logf(s);
    float gold = 0.0f;
    int prev = 0;
    for (int t = 0; t < len; ++t) {
      int tg = tags[b * Ssz + t];
      gold += emit[((size_t)b * Ssz + t) * ELD + tg];
      if (t >= 1) gold += tr[prev * NTAGS + tg];
      prev = tg;
    }
    outv[b] = pred - gold;
  }
}

// ============================ launch ============================
#define CDIV(a, b) (((a) + (b) - 1) / (b))

extern "C" void kernel_launch(void* const* d_in, const int* in_sizes, int n_in,
                              void* d_out, int out_size, void* d_ws, size_t ws_size,
                              hipStream_t stream) {
  const int*   sentences = (const int*)d_in[0];
  const int*   lengths   = (const int*)d_in[1];
  const int*   words     = (const int*)d_in[2];
  const int*   tags      = (const int*)d_in[3];
  const float* char_emb  = (const float*)d_in[4];
  const float* cWi_f = (const float*)d_in[5];
  const float* cWh_f = (const float*)d_in[6];
  const float* cb_f  = (const float*)d_in[7];
  const float* cWi_b = (const float*)d_in[8];
  const float* cWh_b = (const float*)d_in[9];
  const float* cb_b  = (const float*)d_in[10];
  const float* word_emb = (const float*)d_in[11];
  const float* wWi_f = (const float*)d_in[12];
  const float* wWh_f = (const float*)d_in[13];
  const float* wb_f  = (const float*)d_in[14];
  const float* wWi_b = (const float*)d_in[15];
  const float* wWh_b = (const float*)d_in[16];
  const float* wb_b  = (const float*)d_in[17];
  const float* We    = (const float*)d_in[18];
  const float* be    = (const float*)d_in[19];
  const float* trans = (const float*)d_in[20];

  // ---- scratch layout (~135 MB) ----
  char* ws = (char*)d_ws;
  size_t off = 0;
  auto alloc = [&](size_t bytes) -> void* {
    void* p = ws + off;
    off += (bytes + 255) & ~(size_t)255;
    return p;
  };
  __bf16* ce      = (__bf16*)alloc((size_t)NCHAR * CKP * 2);
  __bf16* cWifP   = (__bf16*)alloc((size_t)CG4 * CKP * 2);
  __bf16* cWibP   = (__bf16*)alloc((size_t)CG4 * CKP * 2);
  __bf16* cWhfP   = (__bf16*)alloc((size_t)CG4 * CHKP * 2);
  __bf16* cWhbP   = (__bf16*)alloc((size_t)CG4 * CHKP * 2);
  float*  cbfP    = (float*)alloc((size_t)CG4 * 4);
  float*  cbbP    = (float*)alloc((size_t)CG4 * 4);
  float*  hf      = (float*)alloc((size_t)NSEQ * CHID * 4);
  float*  hb      = (float*)alloc((size_t)NSEQ * CHID * 4);
  __bf16* x       = (__bf16*)alloc((size_t)NSEQ * XK * 2);
  __bf16* xr      = (__bf16*)alloc((size_t)NSEQ * XK * 2);
  __bf16* wWifP   = (__bf16*)alloc((size_t)WG4 * XK * 2);
  __bf16* wWibP   = (__bf16*)alloc((size_t)WG4 * XK * 2);
  __bf16* wWhfP   = (__bf16*)alloc((size_t)WG4 * HID * 2);
  __bf16* wWhbP   = (__bf16*)alloc((size_t)WG4 * HID * 2);
  float*  wbfP    = (float*)alloc((size_t)WG4 * 4);
  float*  wbbP    = (float*)alloc((size_t)WG4 * 4);
  float*  xWi_f   = (float*)alloc((size_t)NSEQ * WG4 * 4);
  float*  xWi_b   = (float*)alloc((size_t)NSEQ * WG4 * 4);
  float*  h_fwd   = (float*)alloc((size_t)NSEQ * HID * 4);
  float*  hr_all  = (float*)alloc((size_t)NSEQ * HID * 4);
  __bf16* WeP     = (__bf16*)alloc((size_t)64 * 2 * HID * 2);
  __bf16* outb    = (__bf16*)alloc((size_t)NSEQ * 2 * HID * 2);
  float*  emit    = (float*)alloc((size_t)NSEQ * ELD * 4);
  (void)ws_size; (void)in_sizes; (void)n_in; (void)out_size;

  // ---- stage 0: weight prep (bf16, K-pad, gate-interleave) + ce gather ----
  conv_gate_w_kernel<<<CDIV((long)CG4 * CKP, 256), 256, 0, stream>>>(cWi_f, cWifP, CHID, CEMBD, CKP);
  conv_gate_w_kernel<<<CDIV((long)CG4 * CKP, 256), 256, 0, stream>>>(cWi_b, cWibP, CHID, CEMBD, CKP);
  conv_gate_w_kernel<<<CDIV((long)CG4 * CHKP, 256), 256, 0, stream>>>(cWh_f, cWhfP, CHID, CHID, CHKP);
  conv_gate_w_kernel<<<CDIV((long)CG4 * CHKP, 256), 256, 0, stream>>>(cWh_b, cWhbP, CHID, CHID, CHKP);
  conv_gate_w_kernel<<<CDIV((long)WG4 * XK, 256), 256, 0, stream>>>(wWi_f, wWifP, HID, EMBD + CHID, XK);
  conv_gate_w_kernel<<<CDIV((long)WG4 * XK, 256), 256, 0, stream>>>(wWi_b, wWibP, HID, EMBD + CHID, XK);
  conv_gate_w_kernel<<<CDIV((long)WG4 * HID, 256), 256, 0, stream>>>(wWh_f, wWhfP, HID, HID, HID);
  conv_gate_w_kernel<<<CDIV((long)WG4 * HID, 256), 256, 0, stream>>>(wWh_b, wWhbP, HID, HID, HID);
  conv_plain_w_kernel<<<CDIV((long)64 * 2 * HID, 256), 256, 0, stream>>>(We, WeP, NTAGS, 64, 2 * HID);
  perm_bias_kernel<<<CDIV(CG4, 256), 256, 0, stream>>>(cb_f, cbfP, CHID);
  perm_bias_kernel<<<CDIV(CG4, 256), 256, 0, stream>>>(cb_b, cbbP, CHID);
  perm_bias_kernel<<<CDIV(WG4, 256), 256, 0, stream>>>(wb_f, wbfP, HID);
  perm_bias_kernel<<<CDIV(WG4, 256), 256, 0, stream>>>(wb_b, wbbP, HID);
  ce_gather_kernel<<<CDIV((long)NCHAR * CKP, 256), 256, 0, stream>>>(words, char_emb, ce);

  // ---- stage 1: char BiLSTM (persistent WMMA recurrence) ----
  char_lstm_kernel<<<NSEQ / 16, 128, 0, stream>>>(ce, cWifP, cWhfP, cbfP, hf, 0);
  char_lstm_kernel<<<NSEQ / 16, 128, 0, stream>>>(ce, cWibP, cWhbP, cbbP, hb, 1);

  // ---- stage 2: x / xr assembly ----
  x_assembly_kernel<<<CDIV((long)NSEQ * XK, 256), 256, 0, stream>>>(sentences, lengths, word_emb,
                                                                    hf, hb, x, xr);

  // ---- stage 3: word input projections (big WMMA GEMMs: 8192x1024x416) ----
  gemm_bf16_kernel<<<dim3(NSEQ / 16, WG4 / 16 / 8), 256, 0, stream>>>(
      x, wWifP, xWi_f, nullptr, 0, XK, WG4, XK / 32, WG4 / 16);
  gemm_bf16_kernel<<<dim3(NSEQ / 16, WG4 / 16 / 8), 256, 0, stream>>>(
      xr, wWibP, xWi_b, nullptr, 0, XK, WG4, XK / 32, WG4 / 16);

  // ---- stage 4: word BiLSTM (persistent WMMA recurrence) ----
  word_lstm_kernel<<<Bsz / 16, 256, 0, stream>>>(xWi_f, wWhfP, wbfP, h_fwd);
  word_lstm_kernel<<<Bsz / 16, 256, 0, stream>>>(xWi_b, wWhbP, wbbP, hr_all);

  // ---- stage 5: emission GEMM (8192x64x512) + CRF ----
  out_assembly_kernel<<<CDIV((long)NSEQ * 2 * HID, 256), 256, 0, stream>>>(lengths, h_fwd, hr_all, outb);
  gemm_bf16_kernel<<<dim3(NSEQ / 16, 1), 128, 0, stream>>>(
      outb, WeP, emit, be, NTAGS, 2 * HID, ELD, (2 * HID) / 32, ELD / 16);
  crf_kernel<<<Bsz, 64, 0, stream>>>(emit, lengths, tags, trans, (float*)d_out);
}
